// RGATClassic_1013612282534
// MI455X (gfx1250) — matrix operand order repaired and checked by reading, not compiled, not run on hardware
//
#include <hip/hip_runtime.h>
#include <hip/hip_bf16.h>

// ---------------------------------------------------------------------------
// RGATConv (heads=1, additive attention, across-relation) for two graphs.
// N=50000 nodes, E=800000 edges, IN=HID=128, R=8.
// Dominant FLOPs: per-relation transform xt[r] = X @ W_r  -> f16 WMMA
// (16x16x32, f32 accumulate), B pre-staged in fragment layout, A row-major in 16-byte chunks, so every
// fragment load is a coalesced b128 vector load; B double-buffered across
// n-tiles so WMMAs overlap the next tile's loads.
// Everything else is scatter/gather with atomics (bandwidth-bound).
// ---------------------------------------------------------------------------

#define NN   50000
#define EE   800000
#define INC  128
#define HIDC 128
#define RREL 8
#define NEGS 0.2f

typedef __attribute__((ext_vector_type(16))) _Float16 v16h;
typedef __attribute__((ext_vector_type(8)))  _Float16 v8h;
typedef __attribute__((ext_vector_type(8)))  float    v8f;

// ------------------------------- helpers -----------------------------------

__device__ __forceinline__ unsigned flip_f32(float f) {
    unsigned u = __float_as_uint(f);
    return (u & 0x80000000u) ? ~u : (u | 0x80000000u);
}
__device__ __forceinline__ float unflip_f32(unsigned u) {
    return (u & 0x80000000u) ? __uint_as_float(u ^ 0x80000000u)
                             : __uint_as_float(~u);
}

// ------------------------------ kernels ------------------------------------

// Generic f32 -> f16 conversion (used for x).
__global__ __launch_bounds__(256)
void k_f32_to_f16(const float* __restrict__ src, _Float16* __restrict__ dst, int n) {
    int i = blockIdx.x * blockDim.x + threadIdx.x;
    if (i < n) dst[i] = (_Float16)src[i];
}

// Stage W (f32, [R][128][128]) into WMMA B-fragment layout:
//   WhB[(((r*8 + nt)*4 + kb)*32 + lane)*16 + i]  =  W[r][krow][col]
// where krow = kb*32 + (i>>3)*16 + (lane>>4)*8 + (i&7), col = nt*16 + (lane&15).
// Each lane's v16h fragment becomes one contiguous 32-byte chunk.
__global__ __launch_bounds__(256)
void k_stageB(const float* __restrict__ W, _Float16* __restrict__ WhB) {
    int idx = blockIdx.x * blockDim.x + threadIdx.x;
    if (idx >= RREL * INC * HIDC) return;
    int i    = idx & 15;
    int lane = (idx >> 4) & 31;
    int kb   = (idx >> 9) & 3;
    int nt   = (idx >> 11) & 7;
    int r    = idx >> 14;
    int krow = kb * 32 + (i >> 3) * 16 + (lane >> 4) * 8 + (i & 7);
    int col  = nt * 16 + (lane & 15);
    WhB[idx] = (_Float16)W[((size_t)r * INC + krow) * HIDC + col];
}

// Zero the output region for one graph; init amax (flipped uint) and denom.
__global__ __launch_bounds__(256)
void k_init(float* __restrict__ out, unsigned* __restrict__ amaxU,
            float* __restrict__ denom) {
    int i = blockIdx.x * blockDim.x + threadIdx.x;
    if (i < NN * HIDC) out[i] = 0.0f;
    if (i < NN) { amaxU[i] = 0u; denom[i] = 1e-16f; }
}

// wq[r][d] = sum_h W[r][d][h]*q[h] ; wk likewise.  (R*IN threads)
__global__ __launch_bounds__(256)
void k_wqk(const float* __restrict__ W, const float* __restrict__ q,
           const float* __restrict__ k, float* __restrict__ wq,
           float* __restrict__ wk) {
    int idx = blockIdx.x * blockDim.x + threadIdx.x;   // r*IN + d
    if (idx >= RREL * INC) return;
    const float* wrow = W + (size_t)idx * HIDC;
    float sq = 0.f, sk = 0.f;
    #pragma unroll 4
    for (int h = 0; h < HIDC; ++h) { float w = wrow[h]; sq += w * q[h]; sk += w * k[h]; }
    wq[idx] = sq; wk[idx] = sk;
}

// qi[r*N+n] = x[n] . wq[r] ; kj likewise.  (R*N threads)
__global__ __launch_bounds__(256)
void k_qikj(const float* __restrict__ x, const float* __restrict__ wq,
            const float* __restrict__ wk, float* __restrict__ qi,
            float* __restrict__ kj) {
    int idx = blockIdx.x * blockDim.x + threadIdx.x;
    if (idx >= RREL * NN) return;
    int r = idx / NN, n = idx - r * NN;
    const float* xr  = x  + (size_t)n * INC;
    const float* wqr = wq + r * INC;
    const float* wkr = wk + r * INC;
    float sq = 0.f, sk = 0.f;
    #pragma unroll 4
    for (int d = 0; d < INC; ++d) { float xv = xr[d]; sq += xv * wqr[d]; sk += xv * wkr[d]; }
    qi[idx] = sq; kj[idx] = sk;
}

// ---- GEMM helpers ----------------------------------------------------------

__device__ __forceinline__ void load_bfrag(const _Float16* __restrict__ bbase,
                                           int nt, v16h (&dst)[4]) {
    const _Float16* p = bbase + (size_t)nt * (4 * 32 * 16);
    #pragma unroll
    for (int kb = 0; kb < 4; ++kb)
        dst[kb] = *(const v16h*)(p + (size_t)kb * (32 * 16));
}

__device__ __forceinline__ void tile_compute_store(
        const v16h (&a)[4], const v16h (&b)[4],
        _Float16* __restrict__ xt, size_t rowbase, int mrow, int col,
        bool fullTile) {
    v8f c = {};
    #pragma unroll
    for (int kb = 0; kb < 4; ++kb)
        c = __builtin_amdgcn_wmma_f32_16x16x32_f16(
                false, a[kb], false, b[kb], (short)0, c, false, false);
    _Float16* op = xt + rowbase * HIDC + col;
    if (fullTile) {
        #pragma unroll
        for (int i = 0; i < 8; ++i) op[(size_t)i * HIDC] = (_Float16)c[i];
    } else {
        #pragma unroll
        for (int i = 0; i < 8; ++i)
            if (mrow + i < NN) op[(size_t)i * HIDC] = (_Float16)c[i];
    }
}

// xt[r][n][h] = sum_d x[n][d] * W[r][d][h], f16 in, f32 WMMA accum, f16 out.
// Block = 256 threads = 8 waves; block tile = 128 rows x 128 cols for one
// relation; wave tile = 16 rows x 128 cols (8 WMMA n-tiles x 4 k-steps),
// B fragments double-buffered across n-tiles.
__global__ __launch_bounds__(256)
void k_gemm_wmma(const _Float16* __restrict__ xh,    // [N][128]
                 const _Float16* __restrict__ WhB,   // B-fragment layout
                 _Float16* __restrict__ xt)          // [R][N][128]
{
    const int r     = blockIdx.y;
    const int wave  = threadIdx.x >> 5;
    const int lane  = threadIdx.x & 31;
    const int half  = lane >> 4;
    const int lm    = lane & 15;
    const int mbase = blockIdx.x * 128 + wave * 16;

    // ---- A fragments (ISA 16-bit A layout): lane L holds M = L&15;
    // VGPR 0..3 = K(half*8 .. +7), VGPR 4..7 = K(16+half*8 .. +7):
    // two contiguous 16-byte chunks -> b128 vector loads.
    int row  = mbase + lm;
    int rowc = row < NN ? row : 0;
    const _Float16* xrow = xh + (size_t)rowc * INC;
    v16h a[4];
    #pragma unroll
    for (int kb = 0; kb < 4; ++kb) {
        v8h lo = *(const v8h*)(xrow + kb * 32 + half * 8);
        v8h hi = *(const v8h*)(xrow + kb * 32 + 16 + half * 8);
        a[kb] = __builtin_shufflevector(lo, hi, 0, 1, 2, 3, 4, 5, 6, 7,
                                                8, 9, 10, 11, 12, 13, 14, 15);
    }

    // Wave-uniform full-tile predicate: only the final row-tile is partial.
    const bool fullTile =
        (__builtin_amdgcn_readfirstlane(mbase) + 16 <= NN);
    const int    mrow    = mbase + half * 8;
    const size_t rowbase = (size_t)r * NN + mrow;

    const _Float16* bbase = WhB + ((size_t)r * 8 * 4 * 32 + lane) * 16;

    // ---- 8 n-tiles, unrolled by 2 with double-buffered B fragments ----
    v16h bA[4], bB[4];
    load_bfrag(bbase, 0, bA);
    #pragma unroll 1
    for (int nt = 0; nt < 8; nt += 2) {
        load_bfrag(bbase, nt + 1, bB);                 // prefetch odd tile
        tile_compute_store(a, bA, xt, rowbase, mrow, nt * 16 + lm, fullTile);
        if (nt + 2 < 8) load_bfrag(bbase, nt + 2, bA); // prefetch next even
        tile_compute_store(a, bB, xt, rowbase, mrow, (nt + 1) * 16 + lm, fullTile);
    }
}

// Pass A: alpha_e = leaky_relu(qi[r,dst] + kj[r,src]); atomicMax amax[dst].
__global__ __launch_bounds__(256)
void k_edgeA(const int* __restrict__ ei, const int* __restrict__ et,
             const float* __restrict__ qi, const float* __restrict__ kj,
             float* __restrict__ alpha, unsigned* __restrict__ amaxU) {
    int e = blockIdx.x * blockDim.x + threadIdx.x;
    if (e >= EE) return;
    int s = ei[e], d = ei[EE + e], r = et[e];
    float al = qi[r * NN + d] + kj[r * NN + s];
    al = al > 0.f ? al : NEGS * al;
    alpha[e] = al;
    atomicMax(&amaxU[d], flip_f32(al));
}

// Pass B: ex_e = exp(alpha_e - amax[dst]); atomicAdd denom[dst].
__global__ __launch_bounds__(256)
void k_edgeB(const int* __restrict__ ei, const float* __restrict__ alpha,
             const unsigned* __restrict__ amaxU, float* __restrict__ exb,
             float* __restrict__ denom) {
    int e = blockIdx.x * blockDim.x + threadIdx.x;
    if (e >= EE) return;
    int d = ei[EE + e];
    float ex = __expf(alpha[e] - unflip_f32(amaxU[d]));
    exb[e] = ex;
    atomicAdd(&denom[d], ex);
}

// Pass C: out[dst][h] += (ex_e/denom[dst]) * xt[r][src][h].
// 128 threads per edge (2 edges per 256-thread block).
__global__ __launch_bounds__(256)
void k_edgeC(const int* __restrict__ ei, const int* __restrict__ et,
             const float* __restrict__ exb, const float* __restrict__ denom,
             const _Float16* __restrict__ xt, float* __restrict__ out) {
    long long gid = (long long)blockIdx.x * blockDim.x + threadIdx.x;
    int e = (int)(gid >> 7);
    int h = (int)(gid & 127);
    if (e >= EE) return;
    int s = ei[e], d = ei[EE + e], r = et[e];
    float aw = exb[e] / denom[d];
    float v  = (float)xt[(((size_t)r * NN + s) << 7) + h];
    atomicAdd(out + (((size_t)d) << 7) + h, aw * v);
}

// Finalize: out = relu(out + b)
__global__ __launch_bounds__(256)
void k_final(float* __restrict__ out, const float* __restrict__ b) {
    int i = blockIdx.x * blockDim.x + threadIdx.x;
    if (i >= NN * HIDC) return;
    float v = out[i] + b[i & 127];
    out[i] = v > 0.f ? v : 0.f;
}

// ------------------------------ host side ----------------------------------

static inline size_t alg(size_t x) { return (x + 255) & ~(size_t)255; }

extern "C" void kernel_launch(void* const* d_in, const int* in_sizes, int n_in,
                              void* d_out, int out_size, void* d_ws, size_t ws_size,
                              hipStream_t stream) {
    (void)in_sizes; (void)n_in; (void)out_size; (void)ws_size;

    // Workspace carve-out (reused by both graphs; ~122 MB total)
    char* ws = (char*)d_ws;
    size_t o = 0;
    _Float16* xt    = (_Float16*)(ws + o); o += alg((size_t)RREL * NN * HIDC * 2);
    _Float16* xh    = (_Float16*)(ws + o); o += alg((size_t)NN * INC * 2);
    _Float16* WhB   = (_Float16*)(ws + o); o += alg((size_t)RREL * INC * HIDC * 2);
    float*    wq    = (float*)   (ws + o); o += alg((size_t)RREL * INC * 4);
    float*    wk    = (float*)   (ws + o); o += alg((size_t)RREL * INC * 4);
    float*    qi    = (float*)   (ws + o); o += alg((size_t)RREL * NN * 4);
    float*    kj    = (float*)   (ws + o); o += alg((size_t)RREL * NN * 4);
    unsigned* amaxU = (unsigned*)(ws + o); o += alg((size_t)NN * 4);
    float*    denom = (float*)   (ws + o); o += alg((size_t)NN * 4);
    float*    alpha = (float*)   (ws + o); o += alg((size_t)EE * 4);
    float*    exb   = (float*)   (ws + o); o += alg((size_t)EE * 4);

    const int T = 256;
    auto run_graph = [&](const float* x, const int* ei, const int* et,
                         const float* W, const float* q, const float* k,
                         const float* b, float* out) {
        // init output/amax/denom
        k_init<<<(NN * HIDC + T - 1) / T, T, 0, stream>>>(out, amaxU, denom);
        // staging: x -> f16 row-major, W -> f16 B-fragment layout
        k_f32_to_f16<<<(NN * INC + T - 1) / T, T, 0, stream>>>(x, xh, NN * INC);
        k_stageB<<<(RREL * INC * HIDC + T - 1) / T, T, 0, stream>>>(W, WhB);
        // attention projections
        k_wqk<<<(RREL * INC + T - 1) / T, T, 0, stream>>>(W, q, k, wq, wk);
        k_qikj<<<(RREL * NN + T - 1) / T, T, 0, stream>>>(x, wq, wk, qi, kj);
        // per-relation transform via WMMA
        dim3 ggrid((NN + 127) / 128, RREL);
        k_gemm_wmma<<<ggrid, T, 0, stream>>>(xh, WhB, xt);
        // scatter softmax + aggregation
        k_edgeA<<<(EE + T - 1) / T, T, 0, stream>>>(ei, et, qi, kj, alpha, amaxU);
        k_edgeB<<<(EE + T - 1) / T, T, 0, stream>>>(ei, alpha, amaxU, exb, denom);
        long long ctot = (long long)EE * HIDC;
        k_edgeC<<<(unsigned)((ctot + T - 1) / T), T, 0, stream>>>(ei, et, exb, denom, xt, out);
        // bias + relu
        k_final<<<(NN * HIDC + T - 1) / T, T, 0, stream>>>(out, b);
    };

    // d_in order: x1, ei1, et1, x2, ei2, et2, W1, q1, k1, b1, W2, q2, k2, b2
    run_graph((const float*)d_in[0], (const int*)d_in[1], (const int*)d_in[2],
              (const float*)d_in[6], (const float*)d_in[7], (const float*)d_in[8],
              (const float*)d_in[9], (float*)d_out);
    run_graph((const float*)d_in[3], (const int*)d_in[4], (const int*)d_in[5],
              (const float*)d_in[10], (const float*)d_in[11], (const float*)d_in[12],
              (const float*)d_in[13], (float*)d_out + (size_t)NN * HIDC);
}
